// STICKYLlamaAttention_48430051230507
// MI455X (gfx1250) — compile-verified, matrix operand order checked by reading, not run on hardware
//
#include <hip/hip_runtime.h>

// ---------------------------------------------------------------------------
// LLaMA attention layer (prefill) for gfx1250, bf16 WMMA everywhere.
// B=1, S=2048, HID=2048, H=32, KVH=8, HD=64, theta=10000.
// ---------------------------------------------------------------------------

typedef __bf16 bf16_t;
typedef __attribute__((ext_vector_type(16))) __bf16 v16bf;
typedef __attribute__((ext_vector_type(8)))  __bf16 v8bf;
typedef __attribute__((ext_vector_type(8)))  float  v8f;

static constexpr int S_LEN = 2048;
static constexpr int HID   = 2048;
static constexpr int NH    = 32;
static constexpr int NKV   = 8;
static constexpr int HD    = 64;

// ---------------------------------------------------------------------------
// Fragment loader: 16x32 bf16 tile (A operand) or 32x16 tile loaded as rows of
// the [N,K] matrix (B operand).  Per CDNA5 ISA 7.12.2 (16-bit A layout):
//   lanes 0-15  : row = lane,    K = {0..7, 16..23}
//   lanes 16-31 : row = lane-16, K = {8..15, 24..31}
// Two contiguous 8-half (128-bit) runs per lane.
// ---------------------------------------------------------------------------
__device__ __forceinline__ v16bf load_frag_k32(const bf16_t* base, int ld) {
  const int lane = threadIdx.x & 31;
  const int row  = lane & 15;
  const int kb   = (lane >> 4) << 3;     // 0 or 8
  const bf16_t* r = base + (size_t)row * ld;
  v8bf lo = *reinterpret_cast<const v8bf*>(r + kb);
  v8bf hi = *reinterpret_cast<const v8bf*>(r + kb + 16);
  v16bf f;
#pragma unroll
  for (int i = 0; i < 8; ++i) { f[i] = lo[i]; f[i + 8] = hi[i]; }
  return f;
}

// ---------------------------------------------------------------------------
// fp32 -> bf16 conversion
// ---------------------------------------------------------------------------
__global__ void f32_to_bf16_kernel(const float* __restrict__ in,
                                   bf16_t* __restrict__ out, int n) {
  int i = blockIdx.x * blockDim.x + threadIdx.x;
  if (i < n) out[i] = (bf16_t)in[i];
}

// ---------------------------------------------------------------------------
// GEMM: C[M,N] = A[M,K] * B[N,K]^T   (x @ W.T with row-major W = [out,in])
// Block: 256 threads (8 wave32) covering a 32x256 output tile.
// Waves arranged 2 (m-stripes) x 4 (n-strips); each wave owns a 16x64 strip
// = four 16x16 f32 accumulators.
// Double-buffered LDS A slab: next k-slab is fetched to registers during the
// WMMAs and committed to the alternate buffer -> ONE barrier per k-step.
// B strips for the next k-step are prefetched (global_prefetch_b8).
// ---------------------------------------------------------------------------
template <bool OUT_F32>
__global__ void gemm_bf16_wmma(const bf16_t* __restrict__ A,
                               const bf16_t* __restrict__ B,
                               void* __restrict__ Cout,
                               int M, int N, int K) {
  __shared__ unsigned int AsU[2][512];               // 2 x (32x32 bf16)
  const int wave = threadIdx.x >> 5;
  const int lane = threadIdx.x & 31;
  const int mrow = wave >> 2;                        // 0..1
  const int ncol = wave & 3;                         // 0..3
  const int m0 = blockIdx.y * 32;
  const int n0 = blockIdx.x * 256 + ncol * 64;

  // staging addresses: this thread owns dwords t0 and t0+256 of the 512-dword
  // A slab (rows r0 and r0+16, bf16 column c0)
  const int t0 = threadIdx.x;
  const int r0 = t0 >> 4;                            // 16 dwords per row
  const int c0 = (t0 & 15) << 1;
  const bf16_t* a0p = A + (size_t)(m0 + r0) * K + c0;
  const bf16_t* a1p = A + (size_t)(m0 + r0 + 16) * K + c0;

  // prologue: stage slab 0
  unsigned int s0 = *reinterpret_cast<const unsigned int*>(a0p);
  unsigned int s1 = *reinterpret_cast<const unsigned int*>(a1p);
  AsU[0][t0]       = s0;
  AsU[0][t0 + 256] = s1;
  __syncthreads();

  v8f acc[4] = {};
  for (int k0 = 0; k0 < K; k0 += 32) {
    const int cur = (k0 >> 5) & 1;
    const bool more = (k0 + 32) < K;
    if (more) {
      // fetch next A slab into registers (overlaps with WMMAs below)
      s0 = *reinterpret_cast<const unsigned int*>(a0p + k0 + 32);
      s1 = *reinterpret_cast<const unsigned int*>(a1p + k0 + 32);
      // prefetch next B strips (rows n0+lane, n0+32+lane of W, next k-slab)
      __builtin_prefetch(B + (size_t)(n0 + lane) * K + k0 + 32, 0, 1);
      __builtin_prefetch(B + (size_t)(n0 + 32 + lane) * K + k0 + 32, 0, 1);
    }

    v16bf a = load_frag_k32(
        reinterpret_cast<const bf16_t*>(AsU[cur]) + mrow * 16 * 32, 32);
    v16bf b[4];
#pragma unroll
    for (int j = 0; j < 4; ++j)
      b[j] = load_frag_k32(B + (size_t)(n0 + j * 16) * K + k0, K);
#pragma unroll
    for (int j = 0; j < 4; ++j)
      acc[j] = __builtin_amdgcn_wmma_f32_16x16x32_bf16(
          false, a, false, b[j], (short)0, acc[j], false, false);

    if (more) {
      AsU[cur ^ 1][t0]       = s0;
      AsU[cur ^ 1][t0 + 256] = s1;
    }
    __syncthreads();
  }

  // C/D layout: VGPR r -> M = r + 8*(lane>=16), N = lane&15
  const int n  = lane & 15;
  const int mb = (lane >> 4) << 3;
#pragma unroll
  for (int j = 0; j < 4; ++j) {
#pragma unroll
    for (int r = 0; r < 8; ++r) {
      const size_t off =
          (size_t)(m0 + mrow * 16 + mb + r) * N + n0 + j * 16 + n;
      if constexpr (OUT_F32) reinterpret_cast<float*>(Cout)[off] = acc[j][r];
      else                   reinterpret_cast<bf16_t*>(Cout)[off] = (bf16_t)acc[j][r];
    }
  }
}

// ---------------------------------------------------------------------------
// RoPE, in place on bf16 [S, nheads*HD]; one thread per rotation pair.
// ---------------------------------------------------------------------------
__global__ void rope_kernel(bf16_t* __restrict__ X, int nheads, int total) {
  int idx = blockIdx.x * blockDim.x + threadIdx.x;
  if (idx >= total) return;
  const int i  = idx & 31;          // 0..31 (HD/2)
  int rem      = idx >> 5;
  const int hh = rem % nheads;
  const int s  = rem / nheads;
  const float inv_freq = __powf(10000.0f, -(float)(2 * i) / (float)HD);
  const float ang = (float)s * inv_freq;
  float sn, cs;
  __sincosf(ang, &sn, &cs);
  bf16_t* p = X + (size_t)s * nheads * HD + hh * HD;
  const float x1 = (float)p[i];
  const float x2 = (float)p[i + 32];
  p[i]      = (bf16_t)(x1 * cs - x2 * sn);
  p[i + 32] = (bf16_t)(x2 * cs + x1 * sn);
}

// ---------------------------------------------------------------------------
// Causal flash attention, one (head, 64-query tile) per workgroup.
// 8 waves: wave = msub (4 stripes of 16 q-rows) x nhalf (2 halves of 32 cols).
// S = Q K^T via 2 WMMAs (d=64), online softmax in LDS, O += P V via WMMAs
// with V staged transposed in LDS so its B operand is K-contiguous.
// Q fragments are loop-invariant and hoisted out of the key loop.
// ---------------------------------------------------------------------------
__global__ void flash_attn_wmma(const bf16_t* __restrict__ Q,
                                const bf16_t* __restrict__ Kc,
                                const bf16_t* __restrict__ Vc,
                                bf16_t* __restrict__ O) {
  constexpr int BR = 64, BC = 64;
  __shared__ bf16_t Qs[BR][HD];        // 8 KB
  __shared__ float  Ss[BR][BC];        // 16 KB
  __shared__ bf16_t Ps[BR][BC];        // 8 KB
  __shared__ bf16_t Vts[HD][BC];       // 8 KB, Vts[d][key]
  __shared__ float  rowM[BR], rowL[BR], rowA[BR];

  const int h    = blockIdx.y;
  const int qt   = blockIdx.x;
  const int q0   = qt * BR;
  const int kvh  = h >> 2;             // GQA: 4 query heads per kv head
  const int ldq  = NH * HD;            // 2048
  const int ldkv = NKV * HD;           // 512
  const int tid  = threadIdx.x;
  const int wave = tid >> 5, lane = tid & 31;
  const int msub  = wave & 3;
  const int nhalf = wave >> 2;
  const float scale = 0.125f;          // 1/sqrt(64)

  // stage Q tile: 4096 halves, 8 per thread per pass
#pragma unroll
  for (int it = 0; it < 2; ++it) {
    int idx = (tid + it * 256) * 8;
    int r = idx >> 6, c = idx & 63;
    *reinterpret_cast<v8bf*>(&Qs[r][c]) =
        *reinterpret_cast<const v8bf*>(Q + (size_t)(q0 + r) * ldq + h * HD + c);
  }
  if (tid < BR) { rowM[tid] = -1e30f; rowL[tid] = 0.0f; }
  v8f acc0 = {}, acc1 = {};
  __syncthreads();

  // loop-invariant Q fragments (d = 0..31 and 32..63)
  const v16bf qa0 = load_frag_k32(&Qs[msub * 16][0], HD);
  const v16bf qa1 = load_frag_k32(&Qs[msub * 16][32], HD);

  for (int kt = 0; kt <= qt; ++kt) {
    const int k0 = kt * BC;

    // stage V tile transposed: 512 chunks of 8 halves
#pragma unroll
    for (int it = 0; it < 2; ++it) {
      int chunk = tid + it * 256;
      int key = chunk >> 3, d0 = (chunk & 7) << 3;
      v8bf v = *reinterpret_cast<const v8bf*>(
          Vc + (size_t)(k0 + key) * ldkv + kvh * HD + d0);
#pragma unroll
      for (int j = 0; j < 8; ++j) Vts[d0 + j][key] = v[j];
    }

    // scores: this wave's two 16x16 tiles, keys nhalf*32 + {0,16}
#pragma unroll
    for (int nt = 0; nt < 2; ++nt) {
      const int nc0 = nhalf * 32 + nt * 16;
      const bf16_t* kbase = Kc + (size_t)(k0 + nc0) * ldkv + kvh * HD;
      v8f s = {};
      v16bf b0 = load_frag_k32(kbase, ldkv);
      s = __builtin_amdgcn_wmma_f32_16x16x32_bf16(
          false, qa0, false, b0, (short)0, s, false, false);
      v16bf b1 = load_frag_k32(kbase + 32, ldkv);
      s = __builtin_amdgcn_wmma_f32_16x16x32_bf16(
          false, qa1, false, b1, (short)0, s, false, false);
      const int n = lane & 15, mb = (lane >> 4) << 3;
#pragma unroll
      for (int r = 0; r < 8; ++r) {
        const int m = msub * 16 + mb + r;
        float val = s[r] * scale;
        if (k0 + nc0 + n > q0 + m) val = -1e30f;   // causal mask
        Ss[m][nc0 + n] = val;
      }
    }
    __syncthreads();

    // online softmax, one thread per query row
    if (tid < BR) {
      float mOld = rowM[tid], mNew = mOld;
#pragma unroll 8
      for (int c = 0; c < BC; ++c) mNew = fmaxf(mNew, Ss[tid][c]);
      const float alpha = __expf(mOld - mNew);
      float l = rowL[tid] * alpha;
#pragma unroll 8
      for (int c = 0; c < BC; ++c) {
        float p = __expf(Ss[tid][c] - mNew);
        Ps[tid][c] = (bf16_t)p;
        l += p;
      }
      rowM[tid] = mNew; rowL[tid] = l; rowA[tid] = alpha;
    }
    __syncthreads();

    // rescale accumulators, then O += P @ V (4 WMMAs per wave)
    {
      const int mb = (lane >> 4) << 3;
#pragma unroll
      for (int r = 0; r < 8; ++r) {
        const float al = rowA[msub * 16 + mb + r];
        acc0[r] *= al; acc1[r] *= al;
      }
      v16bf p0 = load_frag_k32(&Ps[msub * 16][0], BC);
      v16bf p1 = load_frag_k32(&Ps[msub * 16][32], BC);
      const int d0 = nhalf * 32;
      v16bf va = load_frag_k32(&Vts[d0][0], BC);
      v16bf vb = load_frag_k32(&Vts[d0][32], BC);
      acc0 = __builtin_amdgcn_wmma_f32_16x16x32_bf16(
          false, p0, false, va, (short)0, acc0, false, false);
      acc0 = __builtin_amdgcn_wmma_f32_16x16x32_bf16(
          false, p1, false, vb, (short)0, acc0, false, false);
      v16bf vc = load_frag_k32(&Vts[d0 + 16][0], BC);
      v16bf vd = load_frag_k32(&Vts[d0 + 16][32], BC);
      acc1 = __builtin_amdgcn_wmma_f32_16x16x32_bf16(
          false, p0, false, vc, (short)0, acc1, false, false);
      acc1 = __builtin_amdgcn_wmma_f32_16x16x32_bf16(
          false, p1, false, vd, (short)0, acc1, false, false);
    }
    __syncthreads();
  }

  // normalize and store bf16 output [S, H*HD]
  {
    const int n = lane & 15, mb = (lane >> 4) << 3;
    const int d0 = nhalf * 32;
#pragma unroll
    for (int r = 0; r < 8; ++r) {
      const int m = msub * 16 + mb + r;
      const float inv = 1.0f / rowL[m];
      const size_t base = (size_t)(q0 + m) * ldq + h * HD;
      O[base + d0 + n]      = (bf16_t)(acc0[r] * inv);
      O[base + d0 + 16 + n] = (bf16_t)(acc1[r] * inv);
    }
  }
}

// ---------------------------------------------------------------------------
// Launch pipeline
// ---------------------------------------------------------------------------
extern "C" void kernel_launch(void* const* d_in, const int* in_sizes, int n_in,
                              void* d_out, int out_size, void* d_ws, size_t ws_size,
                              hipStream_t stream) {
  (void)in_sizes; (void)n_in; (void)out_size; (void)ws_size;
  const float* hid = (const float*)d_in[0];
  const float* Wq  = (const float*)d_in[1];
  const float* Wk  = (const float*)d_in[2];
  const float* Wv  = (const float*)d_in[3];
  const float* Wo  = (const float*)d_in[4];

  char* ws = (char*)d_ws;
  size_t off = 0;
  auto wsalloc = [&](size_t elems) {
    bf16_t* p = (bf16_t*)(ws + off);
    off += ((elems * sizeof(bf16_t) + 255) & ~(size_t)255);
    return p;
  };
  bf16_t* hid_b = wsalloc((size_t)S_LEN * HID);
  bf16_t* wq_b  = wsalloc((size_t)HID * HID);
  bf16_t* wk_b  = wsalloc((size_t)NKV * HD * HID);
  bf16_t* wv_b  = wsalloc((size_t)NKV * HD * HID);
  bf16_t* wo_b  = wsalloc((size_t)HID * HID);
  bf16_t* q_b   = wsalloc((size_t)S_LEN * NH * HD);
  bf16_t* k_b   = wsalloc((size_t)S_LEN * NKV * HD);
  bf16_t* v_b   = wsalloc((size_t)S_LEN * NKV * HD);
  bf16_t* ao_b  = wsalloc((size_t)S_LEN * NH * HD);

  auto cvt = [&](const float* src, bf16_t* dst, int n) {
    f32_to_bf16_kernel<<<(n + 255) / 256, 256, 0, stream>>>(src, dst, n);
  };
  cvt(hid, hid_b, S_LEN * HID);
  cvt(Wq,  wq_b,  HID * HID);
  cvt(Wk,  wk_b,  NKV * HD * HID);
  cvt(Wv,  wv_b,  NKV * HD * HID);
  cvt(Wo,  wo_b,  HID * HID);

  const dim3 blk(256);
  // QKV projections (32x256 tile per block)
  gemm_bf16_wmma<false><<<dim3(HID / 256, S_LEN / 32), blk, 0, stream>>>(
      hid_b, wq_b, q_b, S_LEN, HID, HID);
  gemm_bf16_wmma<false><<<dim3((NKV * HD) / 256, S_LEN / 32), blk, 0, stream>>>(
      hid_b, wk_b, k_b, S_LEN, NKV * HD, HID);
  gemm_bf16_wmma<false><<<dim3((NKV * HD) / 256, S_LEN / 32), blk, 0, stream>>>(
      hid_b, wv_b, v_b, S_LEN, NKV * HD, HID);

  // RoPE on Q and K
  {
    int totq = S_LEN * NH * (HD / 2);
    rope_kernel<<<(totq + 255) / 256, 256, 0, stream>>>(q_b, NH, totq);
    int totk = S_LEN * NKV * (HD / 2);
    rope_kernel<<<(totk + 255) / 256, 256, 0, stream>>>(k_b, NKV, totk);
  }

  // causal flash attention
  flash_attn_wmma<<<dim3(S_LEN / 64, NH), blk, 0, stream>>>(q_b, k_b, v_b, ao_b);

  // output projection -> fp32 result
  gemm_bf16_wmma<true><<<dim3(HID / 256, S_LEN / 32), blk, 0, stream>>>(
      ao_b, wo_b, d_out, S_LEN, HID, HID);
}
